// Autoencoder_69106023792984
// MI455X (gfx1250) — matrix operand order repaired and checked by reading, not compiled, not run on hardware
//
#include <hip/hip_runtime.h>
#include <hip/hip_bf16.h>
#include <stdint.h>

// Problem constants (match reference)
#define Vv   2048
#define Ee   512
#define Hh   1024
#define Bb   128
#define Ll   256
#define H3   3072      // 3*H
#define KDd  1536      // E + H  (decoder gi input width / actFC row / fc K)
#define FCWr 2560      // E + 2H (fc_W row width)

typedef __attribute__((ext_vector_type(16))) __bf16 v16bf;
typedef __attribute__((ext_vector_type(8)))  float  v8f;
typedef unsigned int u32x4 __attribute__((ext_vector_type(4)));
typedef int          i32x4 __attribute__((ext_vector_type(4)));
typedef int          i32x8 __attribute__((ext_vector_type(8)));

union FragBF { v16bf v; uint4 q[2]; };

struct GemmDesc {
  const unsigned short* A; int lda;   // bf16 activations, row-major M x K (stride lda)
  const unsigned short* W; int ldw;   // bf16 weights, row-major N x K (stride ldw) -> C = A @ W^T
  float* C; int ldc;
  int M, N, K;
  const float* bias;                  // length N, nullable
  const float* addm; int ldadd;       // [128][ldadd], row = m & 127, nullable
  int out_mode;                       // 0: C[m*ldc+n]; 1: C[((m&127)*Ll + (m>>7)+1)*ldc + n]
};

__device__ __forceinline__ unsigned short f2bf(float f) {
  unsigned u = __float_as_uint(f);
  u += 0x7fffu + ((u >> 16) & 1u);           // round-to-nearest-even
  return (unsigned short)(u >> 16);
}

// ---------------------------------------------------------------------------
// Tensor Data Mover tile fetch: 128 rows x 32 bf16, global->LDS, with HW
// padding of 4 DWORDs after every 16 DWORDs -> 80B LDS row stride.
// ---------------------------------------------------------------------------
#define TILE_K     32
#define LDS_STRIDE 40          // halfs per LDS row (32 + 8 pad)
#define ABUF_H     (128 * LDS_STRIDE)          // 5120 halfs = 10240 B
#define BUF_H      (2 * ABUF_H)                // A + B per buffer, halfs
#define BUF_BYTES  (2 * ABUF_H * 2)            // 20480 B per buffer
#define NBUF       3                           // triple-buffered pipeline
#define SMEM_BYTES (NBUF * BUF_BYTES)

#if defined(__HIP_DEVICE_COMPILE__) && __has_builtin(__builtin_amdgcn_tensor_load_to_lds) && \
    __has_builtin(__builtin_amdgcn_s_wait_tensorcnt)
#define HAVE_TDM 1
#else
#define HAVE_TDM 0
#endif

#if HAVE_TDM
__device__ __forceinline__ void tdm_load_tile(const unsigned short* gbase, int stride_elems,
                                              unsigned lds_byte_off) {
  unsigned long long ga = (unsigned long long)(uintptr_t)gbase;
  // D# group 0: count=1 | lds_addr | global_addr[56:0] | type=2
  u32x4 g0 = { 1u,
               lds_byte_off,
               (unsigned)ga,
               ((unsigned)((ga >> 32) & 0x01FFFFFFull)) | 0x80000000u };
  // D# group 1: data_size=1(2B), pad_enable, pad_interval=3 (16 DW), pad_amount=3 (4 DW)
  //             tensor_dim0=32, tensor_dim1=128, tile 32x128, dim0_stride=ld
  i32x8 g1 = { (int)((1u << 16) | (1u << 20) | (3u << 22) | (3u << 25)),
               (int)(32u << 16),      // tensor_dim0[15:0] << 16 (bits 63:48)
               (int)(128u << 16),     // tensor_dim0 hi = 0 | tensor_dim1 lo = 128
               (int)(32u << 16),      // tensor_dim1 hi = 0 | tile_dim0 = 32
               128,                   // tile_dim1 = 128, tile_dim2 = 0
               stride_elems,          // tensor_dim0_stride (element units)
               0, 0 };
  i32x4 z4 = {0, 0, 0, 0};
#if __clang_major__ >= 23
  i32x8 z8 = {0, 0, 0, 0, 0, 0, 0, 0};
  __builtin_amdgcn_tensor_load_to_lds(g0, g1, z4, z4, z8, 0);
#else
  __builtin_amdgcn_tensor_load_to_lds(g0, g1, z4, z4, 0);
#endif
}

__device__ __forceinline__ void tdm_pair(const GemmDesc& d, int bm, int bn, int k0,
                                         unsigned bufoff) {
  tdm_load_tile(d.A + (size_t)bm * d.lda + k0, d.lda, bufoff);
  tdm_load_tile(d.W + (size_t)bn * d.ldw + k0, d.ldw, bufoff + (unsigned)(ABUF_H * 2));
}
#endif

__device__ __forceinline__ void load_fb(FragBF& f, const unsigned short* sB, int bcol, int bko) {
  f.q[0] = *(const uint4*)&sB[bcol * LDS_STRIDE + bko];
  f.q[1] = *(const uint4*)&sB[bcol * LDS_STRIDE + bko + 16];
}

// One 32-deep K-step for this wave. B fragments flow through a rotating
// 3-register pipeline so ds_loads for tile j+2 overlap the WMMA on tile j.
__device__ __forceinline__ void compute_tile(const unsigned short* sA, const unsigned short* sB,
                                             int wave, int lhalf, int hi, v8f acc[8]) {
  const int am  = wave * 16 + lhalf;
  const int alo = hi ? 8 : 0;
  FragBF fa;
  fa.q[0] = *(const uint4*)&sA[am * LDS_STRIDE + alo];
  fa.q[1] = *(const uint4*)&sA[am * LDS_STRIDE + alo + 16];

  const int bko = hi ? 8 : 0;
  FragBF fb[3];
  load_fb(fb[0], sB, lhalf,      bko);
  load_fb(fb[1], sB, lhalf + 16, bko);
#pragma unroll
  for (int j = 0; j < 8; ++j) {
    if (j + 2 < 8) load_fb(fb[(j + 2) % 3], sB, lhalf + (j + 2) * 16, bko);
    acc[j] = __builtin_amdgcn_wmma_f32_16x16x32_bf16(
        false, fa.v, false, fb[j % 3].v, (short)0, acc[j], false, false);
  }
}

// ---------------------------------------------------------------------------
// bf16 WMMA GEMM: 128x128 block tile, 8 waves, K-step 32, TDM triple-buffered
// LDS staging with compile-time buffer offsets (fallback: sync global->LDS).
// ---------------------------------------------------------------------------
__global__ __launch_bounds__(256, 1) void wmma_gemm_dual(GemmDesc d0, GemmDesc d1) {
  GemmDesc d = (blockIdx.z == 0) ? d0 : d1;
  const int bn = blockIdx.x * 128;
  const int bm = blockIdx.y * 128;
  if (bn >= d.N || bm >= d.M) return;   // uniform per block

  extern __shared__ __align__(16) unsigned short smem[];

  const int tid   = threadIdx.x;
  const int wave  = tid >> 5;
  const int lane  = tid & 31;
  const int lhalf = lane & 15;
  const int hi    = lane >> 4;          // 0 or 1

  v8f acc[8];
#pragma unroll
  for (int j = 0; j < 8; ++j) acc[j] = {};

#if HAVE_TDM
  // ---- async-tensor triple-buffered pipeline ----
  const int nk = d.K / TILE_K;
  if (wave == 0) {
    tdm_pair(d, bm, bn, 0, 0u);
    if (nk > 1) {
      tdm_pair(d, bm, bn, TILE_K, BUF_BYTES);
      __builtin_amdgcn_s_wait_tensorcnt(2);   // tile 0 resident, tile 1 in flight
    } else {
      __builtin_amdgcn_s_wait_tensorcnt(0);
    }
  }
  __syncthreads();

  // 3-stage manual unroll: stage s computes buf s, issues DMA into buf (s+2)%3.
  // All LDS offsets are compile-time constants (no div/mod in the loop).
#define GEMM_STAGE(CUR, ISS)                                                        \
  {                                                                                 \
    if (wave == 0 && it + 2 < nk)                                                   \
      tdm_pair(d, bm, bn, (it + 2) * TILE_K, (unsigned)((ISS) * BUF_BYTES));        \
    compute_tile(smem + (CUR) * BUF_H, smem + (CUR) * BUF_H + ABUF_H,               \
                 wave, lhalf, hi, acc);                                             \
    if (wave == 0) {                                                                \
      if (it + 2 < nk) __builtin_amdgcn_s_wait_tensorcnt(2);                        \
      else             __builtin_amdgcn_s_wait_tensorcnt(0);                        \
    }                                                                               \
    __syncthreads();                                                                \
    ++it;                                                                           \
  }

  int it = 0;
  while (it < nk) {
    GEMM_STAGE(0, 2);
    if (it < nk) GEMM_STAGE(1, 0);
    if (it < nk) GEMM_STAGE(2, 1);
  }
#undef GEMM_STAGE
#else
  // ---- fallback: synchronous global->LDS staging (single buffer) ----
  const int arow  = tid >> 1;           // 0..127
  const int apart = (tid & 1) * 16;     // half offset 0 or 16
  const unsigned short* gA = d.A + (size_t)(bm + arow) * d.lda + apart;
  const unsigned short* gW = d.W + (size_t)(bn + arow) * d.ldw + apart;

  for (int k0 = 0; k0 < d.K; k0 += TILE_K) {
    uint4 av0 = *(const uint4*)(gA + k0);
    uint4 av1 = *(const uint4*)(gA + k0 + 8);
    uint4 bv0 = *(const uint4*)(gW + k0);
    uint4 bv1 = *(const uint4*)(gW + k0 + 8);
    *(uint4*)&smem[arow * LDS_STRIDE + apart]              = av0;
    *(uint4*)&smem[arow * LDS_STRIDE + apart + 8]          = av1;
    *(uint4*)&smem[ABUF_H + arow * LDS_STRIDE + apart]     = bv0;
    *(uint4*)&smem[ABUF_H + arow * LDS_STRIDE + apart + 8] = bv1;
    __syncthreads();
    if (k0 + TILE_K < d.K) {
      __builtin_prefetch(gA + k0 + TILE_K, 0, 1);
      __builtin_prefetch(gW + k0 + TILE_K, 0, 1);
    }
    compute_tile(smem, smem + ABUF_H, wave, lhalf, hi, acc);
    __syncthreads();
  }
#endif

  // Epilogue: C/D layout -> VGPR i: M = i + 8*hi, N = lane&15
  const int mbase = bm + wave * 16 + hi * 8;
#pragma unroll
  for (int j = 0; j < 8; ++j) {
    const int n = bn + j * 16 + lhalf;
    float badd = d.bias ? d.bias[n] : 0.0f;
#pragma unroll
    for (int i = 0; i < 8; ++i) {
      const int m = mbase + i;
      float v = acc[j][i] + badd;
      if (d.addm) v += d.addm[(size_t)(m & 127) * d.ldadd + n];
      size_t off;
      if (d.out_mode == 0) off = (size_t)m * d.ldc + n;
      else                 off = ((size_t)(m & 127) * Ll + (m >> 7) + 1) * (size_t)d.ldc + n;
      d.C[off] = v;
    }
  }
}

// ---------------------------------------------------------------------------
// Elementwise / data-movement kernels
// ---------------------------------------------------------------------------
__global__ void cvt_f32_bf16(const float* __restrict__ s, unsigned short* __restrict__ d, int n) {
  int i = blockIdx.x * blockDim.x + threadIdx.x;
  if (i < n) d[i] = f2bf(s[i]);
}

// x_emb[(t*B+b)*E + e] = bf16(emb_enc[seq[b*L+t]*E + e])
__global__ void gather_enc(const int* __restrict__ seq, const float* __restrict__ tab,
                           unsigned short* __restrict__ out) {
  int idx = blockIdx.x * blockDim.x + threadIdx.x;      // < L*B*E
  int e = idx & (Ee - 1);
  int r = idx >> 9;                                     // t*B + b
  int t = r >> 7, b = r & (Bb - 1);
  int tok = seq[b * Ll + t];
  out[(size_t)r * Ee + e] = f2bf(tab[(size_t)tok * Ee + e]);
}

// actFC[(t*B+b)*KD + e] = bf16(emb_dec[seq[b*L+t]*E + e]),  t in 0..L-2
__global__ void gather_dec(const int* __restrict__ seq, const float* __restrict__ tab,
                           unsigned short* __restrict__ out) {
  int idx = blockIdx.x * blockDim.x + threadIdx.x;      // < (L-1)*B*E
  int e = idx & (Ee - 1);
  int r = idx >> 9;
  int t = r >> 7, b = r & (Bb - 1);
  int tok = seq[b * Ll + t];
  out[(size_t)r * KDd + e] = f2bf(tab[(size_t)tok * Ee + e]);
}

__global__ void gru_gates(const float* __restrict__ gi, const float* __restrict__ gh,
                          const float* __restrict__ b_ih, const float* __restrict__ b_hh,
                          float* __restrict__ h, unsigned short* __restrict__ h_bf,
                          unsigned short* __restrict__ hcopy, int ldcopy) {
  int idx = blockIdx.x * blockDim.x + threadIdx.x;      // < B*H
  int b = idx >> 10, j = idx & (Hh - 1);
  int g = b * H3 + j;
  float r  = 1.0f / (1.0f + __expf(-((gi[g] + b_ih[j]) + (gh[g] + b_hh[j]))));
  float z  = 1.0f / (1.0f + __expf(-((gi[g + Hh] + b_ih[j + Hh]) + (gh[g + Hh] + b_hh[j + Hh]))));
  float nn = tanhf((gi[g + 2 * Hh] + b_ih[j + 2 * Hh]) + r * (gh[g + 2 * Hh] + b_hh[j + 2 * Hh]));
  float hv = (1.0f - z) * nn + z * h[idx];
  h[idx] = hv;
  unsigned short hb = f2bf(hv);
  h_bf[idx] = hb;
  if (hcopy) hcopy[(size_t)b * ldcopy + j] = hb;
}

__global__ void zero_f32(float* p, int n) {
  int i = blockIdx.x * blockDim.x + threadIdx.x;
  if (i < n) p[i] = 0.0f;
}
__global__ void zero_u16(unsigned short* p, int n) {
  int i = blockIdx.x * blockDim.x + threadIdx.x;
  if (i < n) p[i] = 0;
}
__global__ void zero_out_t0(float* out) {
  int idx = blockIdx.x * blockDim.x + threadIdx.x;      // < B*V
  int b = idx >> 11, v = idx & (Vv - 1);
  out[(size_t)b * Ll * Vv + v] = 0.0f;                  // outputs[b, 0, v]
}
__global__ void copy_f32(const float* __restrict__ s, float* __restrict__ d, int n) {
  int i = blockIdx.x * blockDim.x + threadIdx.x;
  if (i < n) d[i] = s[i];
}

// ---------------------------------------------------------------------------
// Host orchestration
// ---------------------------------------------------------------------------
extern "C" void kernel_launch(void* const* d_in, const int* in_sizes, int n_in,
                              void* d_out, int out_size, void* d_ws, size_t ws_size,
                              hipStream_t stream) {
  (void)in_sizes; (void)n_in; (void)out_size; (void)ws_size;

  const int*   seq     = (const int*)  d_in[0];
  const float* emb_enc = (const float*)d_in[2];
  const float* Wih_e   = (const float*)d_in[3];
  const float* Whh_e   = (const float*)d_in[4];
  const float* bih_e   = (const float*)d_in[5];
  const float* bhh_e   = (const float*)d_in[6];
  const float* emb_dec = (const float*)d_in[7];
  const float* Wih_d   = (const float*)d_in[8];
  const float* Whh_d   = (const float*)d_in[9];
  const float* bih_d   = (const float*)d_in[10];
  const float* bhh_d   = (const float*)d_in[11];
  const float* fcW     = (const float*)d_in[12];
  const float* fcb     = (const float*)d_in[13];
  float* out = (float*)d_out;

  // workspace carve (~168 MB)
  char* p = (char*)d_ws;
  auto carve = [&](size_t bytes) { void* r = (void*)p; p += (bytes + 255) & ~(size_t)255; return r; };
  unsigned short* wihe  = (unsigned short*)carve((size_t)H3 * Ee  * 2);
  unsigned short* whhe  = (unsigned short*)carve((size_t)H3 * Hh  * 2);
  unsigned short* wihd  = (unsigned short*)carve((size_t)H3 * KDd * 2);
  unsigned short* whhd  = (unsigned short*)carve((size_t)H3 * Hh  * 2);
  unsigned short* fcw   = (unsigned short*)carve((size_t)Vv * FCWr * 2);
  unsigned short* xemb  = (unsigned short*)carve((size_t)Ll * Bb * Ee * 2);
  unsigned short* actFC = (unsigned short*)carve((size_t)(Ll - 1) * Bb * KDd * 2);
  float* gi_ws  = (float*)carve((size_t)Bb * H3 * 4);
  float* gh_ws  = (float*)carve((size_t)Bb * H3 * 4);
  float* gi_ctx = (float*)carve((size_t)Bb * H3 * 4);
  float* ctx_fc = (float*)carve((size_t)Bb * Vv * 4);
  float* h_f32  = (float*)carve((size_t)Bb * Hh * 4);
  unsigned short* h_bf = (unsigned short*)carve((size_t)Bb * Hh * 2);

  auto g1 = [](int n) { return dim3((unsigned)((n + 255) / 256)); };

  // 1) weights -> bf16 (live in L2 thereafter)
  cvt_f32_bf16<<<g1(H3 * Ee),   256, 0, stream>>>(Wih_e, wihe, H3 * Ee);
  cvt_f32_bf16<<<g1(H3 * Hh),   256, 0, stream>>>(Whh_e, whhe, H3 * Hh);
  cvt_f32_bf16<<<g1(H3 * KDd),  256, 0, stream>>>(Wih_d, wihd, H3 * KDd);
  cvt_f32_bf16<<<g1(H3 * Hh),   256, 0, stream>>>(Whh_d, whhd, H3 * Hh);
  cvt_f32_bf16<<<g1(Vv * FCWr), 256, 0, stream>>>(fcW,   fcw,  Vv * FCWr);

  // 2) embedding gathers + output/state init
  gather_enc<<<g1(Ll * Bb * Ee),       256, 0, stream>>>(seq, emb_enc, xemb);
  gather_dec<<<g1((Ll - 1) * Bb * Ee), 256, 0, stream>>>(seq, emb_dec, actFC);
  zero_out_t0<<<g1(Bb * Vv), 256, 0, stream>>>(out);
  zero_f32<<<g1(2 * Bb * Hh), 256, 0, stream>>>(out + (size_t)Bb * Ll * Vv + (size_t)Bb * Hh,
                                                2 * Bb * Hh);        // mu, log_var
  zero_f32<<<g1(Bb * Hh), 256, 0, stream>>>(h_f32, Bb * Hh);
  zero_u16<<<g1(Bb * Hh), 256, 0, stream>>>(h_bf, Bb * Hh);

  const dim3 dual_grid(H3 / 128, 1, 2);

  // 3) encoder scan: gi = x_t @ Wih_e^T ; gh = h @ Whh_e^T (dual), then gates
  for (int t = 0; t < Ll; ++t) {
    GemmDesc gi{xemb + (size_t)t * Bb * Ee, Ee, wihe, Ee, gi_ws, H3,
                Bb, H3, Ee, nullptr, nullptr, 0, 0};
    GemmDesc gh{h_bf, Hh, whhe, Hh, gh_ws, H3,
                Bb, H3, Hh, nullptr, nullptr, 0, 0};
    wmma_gemm_dual<<<dual_grid, 256, SMEM_BYTES, stream>>>(gi, gh);
    gru_gates<<<g1(Bb * Hh), 256, 0, stream>>>(gi_ws, gh_ws, bih_e, bhh_e,
                                               h_f32, h_bf, nullptr, 0);
  }

  // 4) context out + loop-invariant decoder terms (dual: N=3072 and N=2048)
  copy_f32<<<g1(Bb * Hh), 256, 0, stream>>>(h_f32, out + (size_t)Bb * Ll * Vv, Bb * Hh);
  {
    GemmDesc gctx{h_bf, Hh, wihd + Ee,       KDd,  gi_ctx, H3,
                  Bb, H3, Hh, nullptr, nullptr, 0, 0};
    GemmDesc gfc {h_bf, Hh, fcw + (Ee + Hh), FCWr, ctx_fc, Vv,
                  Bb, Vv, Hh, nullptr, nullptr, 0, 0};
    wmma_gemm_dual<<<dual_grid, 256, SMEM_BYTES, stream>>>(gctx, gfc);
  }

  // 5) decoder scan: gi = e_t @ Wih_d[:, :E]^T + gi_ctx ; gh = h @ Whh_d^T
  for (int t = 0; t < Ll - 1; ++t) {
    GemmDesc gi{actFC + (size_t)t * Bb * KDd, KDd, wihd, KDd, gi_ws, H3,
                Bb, H3, Ee, nullptr, gi_ctx, H3, 0};
    GemmDesc gh{h_bf, Hh, whhd, Hh, gh_ws, H3,
                Bb, H3, Hh, nullptr, nullptr, 0, 0};
    wmma_gemm_dual<<<dual_grid, 256, SMEM_BYTES, stream>>>(gi, gh);
    gru_gates<<<g1(Bb * Hh), 256, 0, stream>>>(gi_ws, gh_ws, bih_d, bhh_d,
                                               h_f32, h_bf,
                                               actFC + (size_t)t * Bb * KDd + Ee, KDd);
  }

  // 6) batched fc over all timesteps: 32640 x 2048 x 1536, permuted store into
  //    outputs[b, t+1, :], + ctx contribution + fc_b in epilogue
  {
    GemmDesc fc{actFC, KDd, fcw, FCWr, out, Vv,
                (Ll - 1) * Bb, Vv, KDd, fcb, ctx_fc, Vv, 1};
    wmma_gemm_dual<<<dim3(Vv / 128, (Ll - 1), 1), 256, SMEM_BYTES, stream>>>(fc, fc);
  }
}
